// ShawAttentionMQA_70849780515236
// MI455X (gfx1250) — matrix operand order repaired and checked by reading, not compiled, not run on hardware
//
#include <hip/hip_runtime.h>
#include <hip/hip_bf16.h>

// ---------------------------------------------------------------------------
// Shaw relative-position MQA attention for MI455X (gfx1250, wave32, WMMA).
//   q = x@Wq^T  [B,H,S,64]   k,v = x@W{k,v}^T  [B,S,64]  (single KV head)
//   scores = (q·k^T + q·rel[i,j,:]) / 8 ; softmax ; ctx = P@v ; out = ctx@Wo^T+bo
// All contractions via v_wmma_f32_16x16x32_bf16 (f32 accum).
// rel (256 MB) is read exactly once from HBM (block-per-i, shared LDS tile).
// Q staging uses GLOBAL_LOAD_ASYNC_TO_LDS_B128 (builtin confirmed present).
// ---------------------------------------------------------------------------

typedef __attribute__((ext_vector_type(16))) __bf16 v16bf;
typedef __attribute__((ext_vector_type(8)))  float  v8f;
typedef __attribute__((ext_vector_type(4)))  int    v4i;

typedef __attribute__((address_space(1))) v4i gv4i;  // global int4
typedef __attribute__((address_space(3))) v4i lv4i;  // LDS int4

#define S_LEN   1024
#define DM      1024
#define NHEAD   16
#define DH      64
#define NBATCH  4
#define MTOT    4096   // B*S rows

__device__ __forceinline__ v16bf frag_from2(uint4 a, uint4 b) {
  union { uint4 u[2]; v16bf v; } x; x.u[0] = a; x.u[1] = b; return x.v;
}

// A fragment (16x32 bf16, M x K): lane m = lane&15, k-half = lane>>4.
// element e: K = (e&7) + 8*(lane>>4) + 16*(e>>3)  -> two 16B loads per lane.
__device__ __forceinline__ v16bf load_frag_a(const __bf16* tile, int ldk) {
  const int lane = threadIdx.x & 31;
  const int m    = lane & 15;
  const int h8   = (lane >> 4) << 3;
  const __bf16* p = tile + (size_t)m * ldk + h8;
  return frag_from2(*(const uint4*)p, *(const uint4*)(p + 16));
}

// B fragment (32x16 bf16, K x N): lane holds column n = lane&15,
// K = (e&7) + 8*(lane>>4) + 16*(e>>3) -> two 16B loads per lane.
__device__ __forceinline__ v16bf load_frag_b(const __bf16* base, int n0,
                                             int ld, int k0) {
  const int lane = threadIdx.x & 31;
  const __bf16* p = base + (size_t)(n0 + (lane & 15)) * ld + k0 + ((lane >> 4) << 3);
  return frag_from2(*(const uint4*)p, *(const uint4*)(p + 16));
}

__device__ __forceinline__ v8f wmma_bf16(v16bf a, v16bf b, v8f c) {
  return __builtin_amdgcn_wmma_f32_16x16x32_bf16(false, a, false, b,
                                                 (short)0, c, false, false);
}

__device__ __forceinline__ uint4 pack8_bf16(float4 a, float4 b) {
  union { __bf16 h[8]; uint4 u; } p;
  p.h[0] = (__bf16)a.x; p.h[1] = (__bf16)a.y; p.h[2] = (__bf16)a.z; p.h[3] = (__bf16)a.w;
  p.h[4] = (__bf16)b.x; p.h[5] = (__bf16)b.y; p.h[6] = (__bf16)b.z; p.h[7] = (__bf16)b.w;
  return p.u;
}

// ---------------------------------------------------------------------------
// Generic  Y = A[M=4096, K=1024] @ W[N, K]^T  with compile-time epilogue mode.
//   MODE 0: q -> bf16 scatter [s][b*16+h][64]
//   MODE 1: k -> bf16 [m][64]
//   MODE 2: v -> bf16 transposed [b][64][1024]
//   MODE 3: out -> f32 [m][1024] + bias
// Block tile 64x64, 8 waves (4 in M x 2 in N), BK = 32.
// ---------------------------------------------------------------------------
template <int MODE>
__global__ __launch_bounds__(256)
void gemm_xwT(const float* __restrict__ A, const float* __restrict__ W,
              const float* __restrict__ bias, void* __restrict__ outp)
{
  __shared__ __align__(16) __bf16 As[64 * 32];
  __shared__ __align__(16) __bf16 Bs[64 * 32];

  const int t    = threadIdx.x;
  const int lane = t & 31;
  const int w    = t >> 5;
  const int gm0  = blockIdx.x * 64;
  const int gn0  = blockIdx.y * 64;
  const int wm   = (w & 3) * 16;
  const int wn   = (w >> 2) * 32;

  const int row  = t >> 2;          // 0..63
  const int col  = (t & 3) * 8;     // 0,8,16,24

  v8f acc0 = {}; v8f acc1 = {};

  for (int kt = 0; kt < DM; kt += 32) {
    const float* ap = A + (size_t)(gm0 + row) * DM + kt + col;
    float4 a0 = *(const float4*)ap;
    float4 a1 = *(const float4*)(ap + 4);
    *(uint4*)&As[row * 32 + col] = pack8_bf16(a0, a1);

    const float* bp = W + (size_t)(gn0 + row) * DM + kt + col;
    float4 b0 = *(const float4*)bp;
    float4 b1 = *(const float4*)(bp + 4);
    *(uint4*)&Bs[row * 32 + col] = pack8_bf16(b0, b1);

    __syncthreads();

    v16bf af = load_frag_a(As + wm * 32, 32);
    v16bf f0 = load_frag_b(Bs, wn, 32, 0);
    v16bf f1 = load_frag_b(Bs, wn + 16, 32, 0);
    acc0 = wmma_bf16(af, f0, acc0);
    acc1 = wmma_bf16(af, f1, acc1);

    __syncthreads();
  }

  const int hi = lane >> 4;
  const int nl = lane & 15;
#pragma unroll
  for (int half = 0; half < 2; ++half) {
    v8f acc = half ? acc1 : acc0;
    const int gn = gn0 + wn + half * 16 + nl;
#pragma unroll
    for (int r = 0; r < 8; ++r) {
      const int gm = gm0 + wm + r + 8 * hi;
      const float v = acc[r];
      if (MODE == 0) {
        const int b = gm >> 10, s = gm & 1023, h = gn >> 6, d = gn & 63;
        ((__bf16*)outp)[(size_t)s * 4096 + (size_t)(b * 16 + h) * 64 + d] = (__bf16)v;
      } else if (MODE == 1) {
        ((__bf16*)outp)[(size_t)gm * 64 + gn] = (__bf16)v;
      } else if (MODE == 2) {
        const int b = gm >> 10, s = gm & 1023;
        ((__bf16*)outp)[((size_t)b * 64 + gn) * 1024 + s] = (__bf16)v;
      } else {
        ((float*)outp)[(size_t)gm * DM + gn] = v + bias[gn];
      }
    }
  }
}

// ---------------------------------------------------------------------------
// Flash attention, one block per query index i.  128 threads = 4 waves,
// wave w handles batch b = w (its 16 rows = the 16 heads).
//   M = 16 heads, K = 64 (d) for scores; c2p batched over (b,h) via shared
//   LDS rel tile.  j tiled by 128, online softmax, P@V via WMMA (vT layout).
// ---------------------------------------------------------------------------
__global__ __launch_bounds__(128)
void attn_kernel(const __bf16* __restrict__ qb,   // [S][64 bh][64 d]
                 const __bf16* __restrict__ kb,   // [B*S][64]
                 const __bf16* __restrict__ vbT,  // [B][64 d][S]
                 const float*  __restrict__ rel,  // [S][S][64] f32
                 float* __restrict__ ctx)         // [B][S][1024]
{
  __shared__ __align__(16) __bf16 Qs[64 * 64];     // 8 KB
  __shared__ __align__(16) __bf16 RelT[128 * 64];  // 16 KB
  __shared__ __align__(16) __bf16 Pw[4 * 16 * 128];// 16 KB (wave-private)

  const int i    = blockIdx.x;
  const int t    = threadIdx.x;
  const int lane = t & 31;
  const int w    = t >> 5;      // batch b
  const int nl   = lane & 15;
  const int hi   = lane >> 4;

  // ---- stage Q_all[i] (contiguous 8 KB, already bf16) via async DMA to LDS
#if defined(__has_builtin) && __has_builtin(__builtin_amdgcn_global_load_async_to_lds_b128)
  {
    const char* gsrc = (const char*)(qb + (size_t)i * 4096);
    char* lQs = (char*)Qs;
#pragma unroll
    for (int c = 0; c < 4; ++c) {
      __builtin_amdgcn_global_load_async_to_lds_b128(
          (gv4i*)(gsrc + (size_t)(t * 4 + c) * 16),
          (lv4i*)(lQs + (size_t)(t * 4 + c) * 16),
          0, 0);
    }
#if __has_builtin(__builtin_amdgcn_s_wait_asynccnt)
    __builtin_amdgcn_s_wait_asynccnt(0);
#else
    asm volatile("s_wait_asynccnt 0x0" ::: "memory");
#endif
  }
#else
  {
    const uint4* src = (const uint4*)(qb + (size_t)i * 4096);
    uint4* dst = (uint4*)Qs;
#pragma unroll
    for (int c = 0; c < 4; ++c) dst[t * 4 + c] = src[t * 4 + c];
  }
#endif
  __syncthreads();

  const v16bf aq0 = load_frag_a(Qs + w * 1024, 64);       // d = 0..31
  const v16bf aq1 = load_frag_a(Qs + w * 1024 + 32, 64);  // d = 32..63

  float mrow[8], lrow[8];
#pragma unroll
  for (int r = 0; r < 8; ++r) { mrow[r] = -1e30f; lrow[r] = 0.f; }
  v8f O[4] = {};

  const __bf16* kbase = kb  + (size_t)w * S_LEN * 64;
  const __bf16* vbase = vbT + (size_t)w * 64 * S_LEN;
  __bf16* pw = Pw + w * 16 * 128;

  for (int jt = 0; jt < S_LEN; jt += 128) {
    __syncthreads();  // previous tile's RelT readers done
    { // stage rel[i, jt:jt+64, :] f32 -> bf16 (read once from HBM)
      const int jl = t >> 1;
      const int d0 = (t & 1) * 32;
      const float* rp = rel + ((size_t)i * S_LEN + jt + jl) * 64 + d0;
#pragma unroll
      for (int c = 0; c < 4; ++c) {
        float4 f0 = *(const float4*)(rp + c * 8);
        float4 f1 = *(const float4*)(rp + c * 8 + 4);
        *(uint4*)&RelT[jl * 64 + d0 + c * 8] = pack8_bf16(f0, f1);
      }
      if (jt + 128 < S_LEN)  // global_prefetch_b8 of next tile
        __builtin_prefetch(rel + ((size_t)i * S_LEN + jt + 128) * 64 + t * 64, 0, 1);
    }
    __syncthreads();

    // ---- scores: c2c (per-b K) + c2p (shared rel), K=64 -> 2 WMMA each ----
    v8f c[8];
#pragma unroll
    for (int nt = 0; nt < 8; ++nt) {
      v8f cc = {};
      v16bf bk0 = load_frag_b(kbase, jt + nt * 16, 64, 0);
      v16bf bk1 = load_frag_b(kbase, jt + nt * 16, 64, 32);
      cc = wmma_bf16(aq0, bk0, cc);
      cc = wmma_bf16(aq1, bk1, cc);
      v16bf br0 = load_frag_b(RelT, nt * 16, 64, 0);
      v16bf br1 = load_frag_b(RelT, nt * 16, 64, 32);
      cc = wmma_bf16(aq0, br0, cc);
      cc = wmma_bf16(aq1, br1, cc);
      c[nt] = cc;
    }

    // ---- online softmax per row (r + 8*hi), 16-lane shfl reductions ----
#pragma unroll
    for (int r = 0; r < 8; ++r) {
      float tm = -1e30f;
#pragma unroll
      for (int nt = 0; nt < 8; ++nt) {
        c[nt][r] *= 0.125f;   // 1/sqrt(64)
        tm = fmaxf(tm, c[nt][r]);
      }
      tm = fmaxf(tm, __shfl_xor(tm, 1));
      tm = fmaxf(tm, __shfl_xor(tm, 2));
      tm = fmaxf(tm, __shfl_xor(tm, 4));
      tm = fmaxf(tm, __shfl_xor(tm, 8));
      const float mnew = fmaxf(mrow[r], tm);
      const float fac  = __expf(mrow[r] - mnew);
      mrow[r] = mnew;
      float ts = 0.f;
#pragma unroll
      for (int nt = 0; nt < 8; ++nt) {
        const float p = __expf(c[nt][r] - mnew);
        c[nt][r] = p;
        ts += p;
      }
      ts += __shfl_xor(ts, 1);
      ts += __shfl_xor(ts, 2);
      ts += __shfl_xor(ts, 4);
      ts += __shfl_xor(ts, 8);
      lrow[r] = lrow[r] * fac + ts;
#pragma unroll
      for (int dn = 0; dn < 4; ++dn) O[dn][r] *= fac;
    }

    // ---- P (bf16) to wave-private LDS, then P(16x128) @ V(128x64) ----
#pragma unroll
    for (int r = 0; r < 8; ++r)
#pragma unroll
      for (int nt = 0; nt < 8; ++nt)
        pw[(r + 8 * hi) * 128 + nt * 16 + nl] = (__bf16)c[nt][r];

    v16bf ap[4];
#pragma unroll
    for (int kk = 0; kk < 4; ++kk) ap[kk] = load_frag_a(pw + kk * 32, 128);
#pragma unroll
    for (int dn = 0; dn < 4; ++dn) {
#pragma unroll
      for (int kk = 0; kk < 4; ++kk) {
        v16bf bv = load_frag_b(vbase, dn * 16, S_LEN, jt + kk * 32);
        O[dn] = wmma_bf16(ap[kk], bv, O[dn]);
      }
    }
  }

  // ---- normalize, write ctx[b][i][h*64+d] ----
#pragma unroll
  for (int dn = 0; dn < 4; ++dn) {
#pragma unroll
    for (int r = 0; r < 8; ++r) {
      const int h = r + 8 * hi;
      ctx[((size_t)w * S_LEN + i) * DM + h * 64 + dn * 16 + nl] =
          O[dn][r] / lrow[r];
    }
  }
}

// ---------------------------------------------------------------------------
extern "C" void kernel_launch(void* const* d_in, const int* in_sizes, int n_in,
                              void* d_out, int out_size, void* d_ws, size_t ws_size,
                              hipStream_t stream) {
  (void)in_sizes; (void)n_in; (void)out_size; (void)ws_size;
  const float* x   = (const float*)d_in[0];
  const float* rel = (const float*)d_in[1];
  const float* Wq  = (const float*)d_in[2];
  const float* Wk  = (const float*)d_in[3];
  const float* Wv  = (const float*)d_in[4];
  const float* Wo  = (const float*)d_in[5];
  const float* bo  = (const float*)d_in[6];
  float* out = (float*)d_out;

  char* ws = (char*)d_ws;
  __bf16* qb  = (__bf16*)(ws);                                     // 8 MB
  __bf16* kbp = (__bf16*)(ws + (size_t)(8u << 20));                // 512 KB
  __bf16* vbT = (__bf16*)(ws + (size_t)(8u << 20) + (512u << 10)); // 512 KB
  float*  ctx = (float*)(ws + (size_t)(9u << 20));                 // 16 MB

  gemm_xwT<0><<<dim3(64, 16), dim3(256), 0, stream>>>(x, Wq, nullptr, qb);
  gemm_xwT<1><<<dim3(64, 1),  dim3(256), 0, stream>>>(x, Wk, nullptr, kbp);
  gemm_xwT<2><<<dim3(64, 1),  dim3(256), 0, stream>>>(x, Wv, nullptr, vbT);
  attn_kernel<<<dim3(1024), dim3(128), 0, stream>>>(qb, kbp, vbT, rel, ctx);
  gemm_xwT<3><<<dim3(64, 16), dim3(256), 0, stream>>>(ctx, Wo, bo, out);
}
